// MultiHeadGroupedQueryAttention_34102040330978
// MI455X (gfx1250) — matrix operand order repaired and checked by reading, not compiled, not run on hardware
//
#include <hip/hip_runtime.h>
#include <hip/hip_bf16.h>

#define EMBED 896
#define HQ 16
#define HKV 4
#define HD 56
#define DPAD 64
#define KV_EMBED 224
#define SEQ 2048
#define BATCH 2
#define ROWS (BATCH * SEQ)              // 4096
#define QK_SCALE 0.13363062095621219f   // 1/sqrt(56)
#define LN_EPS 1e-5f

typedef __attribute__((ext_vector_type(16))) _Float16 v16h;
typedef __attribute__((ext_vector_type(8)))  float    v8f;
typedef __attribute__((ext_vector_type(2)))  _Float16 h16x2;

// ---------------- WMMA fragment helpers (CDNA5 wave32 layouts) ----------------
// A matrix 16x32 f16: lane m = l%16, h = l/16; VGPR j holds K pair:
//   j<4 : K = 2j + 8h ; j>=4 : K = 16 + 2(j-4) + 8h     (ISA 7.12.2)
__device__ __forceinline__ int ka_of(int j, int h) {
  return ((j < 4) ? (2 * j) : (16 + 2 * (j - 4))) + 8 * h;
}
// B matrix 32x16 f16: lane n = l%16, h = l/16; VGPR j holds K pair (same as A):
//   j<4 : K = 2j + 8h ; j>=4 : K = 16 + 2(j-4) + 8h
__device__ __forceinline__ int kb_of(int j, int h) {
  return ((j < 4) ? (2 * j) : (16 + 2 * (j - 4))) + 8 * h;
}

__device__ __forceinline__ v8f wmma_f16(v16h a, v16h b, v8f c) {
  return __builtin_amdgcn_wmma_f32_16x16x32_f16(false, a, false, b, (short)0, c,
                                                false, false);
}

// build A fragment from an f32 row (contiguous K), converting to f16
__device__ __forceinline__ v16h afrag_f32(const float* p, int h) {
  v16h f;
#pragma unroll
  for (int j = 0; j < 8; ++j) {
    int k = ka_of(j, h);
    float2 d = *reinterpret_cast<const float2*>(p + k);
    f[2 * j] = (_Float16)d.x;
    f[2 * j + 1] = (_Float16)d.y;
  }
  return f;
}
// build B fragment from an f32 row of W[n][k] (contraction dim contiguous)
__device__ __forceinline__ v16h bfrag_f32(const float* p, int h) {
  v16h f;
#pragma unroll
  for (int j = 0; j < 8; ++j) {
    int k = kb_of(j, h);
    float2 d = *reinterpret_cast<const float2*>(p + k);
    f[2 * j] = (_Float16)d.x;
    f[2 * j + 1] = (_Float16)d.y;
  }
  return f;
}
// f16 A fragment with zero padding past kmax (head dim 56 -> 64)
__device__ __forceinline__ v16h afrag_f16b(const _Float16* p, int h, int koff,
                                           int kmax) {
  v16h f;
#pragma unroll
  for (int j = 0; j < 8; ++j) {
    int k = koff + ka_of(j, h);
    if (k < kmax) {
      h16x2 d = *reinterpret_cast<const h16x2*>(p + k);
      f[2 * j] = d.x;
      f[2 * j + 1] = d.y;
    } else {
      f[2 * j] = (_Float16)0.f;
      f[2 * j + 1] = (_Float16)0.f;
    }
  }
  return f;
}
__device__ __forceinline__ v16h bfrag_f16b(const _Float16* p, int h, int koff,
                                           int kmax) {
  v16h f;
#pragma unroll
  for (int j = 0; j < 8; ++j) {
    int k = koff + kb_of(j, h);
    if (k < kmax) {
      h16x2 d = *reinterpret_cast<const h16x2*>(p + k);
      f[2 * j] = d.x;
      f[2 * j + 1] = d.y;
    } else {
      f[2 * j] = (_Float16)0.f;
      f[2 * j + 1] = (_Float16)0.f;
    }
  }
  return f;
}

// ---------------- GEMM: C(M,N) = A(M,K) x B(N,K)^T ----------------
// 32 x (16*NT) tile per wave (2 row tiles share each B fragment).
// MODE 0: f16 row-major out; MODE 1: f32 row-major out;
// MODE 2: f16 transposed+padded out -> Vt[b][hk][DPAD][SEQ]
template <int MODE, int NT>
__global__ __launch_bounds__(32) void gemm16(const float* __restrict__ A,
                                             const float* __restrict__ B,
                                             void* __restrict__ C, int M, int N,
                                             int K) {
  const int lane = threadIdx.x;
  const int lm = lane & 15, lh = lane >> 4;
  const int row0 = blockIdx.y * 32;
  const int col0 = blockIdx.x * (16 * NT);

  v8f zero = {};
  v8f acc[2][NT];
#pragma unroll
  for (int mt = 0; mt < 2; ++mt)
#pragma unroll
    for (int t = 0; t < NT; ++t) acc[mt][t] = zero;

  const float* arow0 = A + (size_t)(row0 + lm) * K;
  const float* arow1 = A + (size_t)(row0 + 16 + lm) * K;
  for (int k0 = 0; k0 < K; k0 += 32) {
    __builtin_prefetch(arow0 + k0 + 128, 0, 1);  // global_prefetch_b8
    __builtin_prefetch(arow1 + k0 + 128, 0, 1);
    v16h a0 = afrag_f32(arow0 + k0, lh);
    v16h a1 = afrag_f32(arow1 + k0, lh);
#pragma unroll
    for (int t = 0; t < NT; ++t) {
      const float* brow = B + (size_t)(col0 + t * 16 + lm) * K + k0;
      v16h bf = bfrag_f32(brow, lh);
      acc[0][t] = wmma_f16(a0, bf, acc[0][t]);
      acc[1][t] = wmma_f16(a1, bf, acc[1][t]);
    }
  }
#pragma unroll
  for (int mt = 0; mt < 2; ++mt) {
#pragma unroll
    for (int t = 0; t < NT; ++t) {
#pragma unroll
      for (int r = 0; r < 8; ++r) {
        int row = row0 + mt * 16 + r + 8 * lh;
        int col = col0 + t * 16 + lm;
        float v = acc[mt][t][r];
        if (MODE == 0) {
          ((_Float16*)C)[(size_t)row * N + col] = (_Float16)v;
        } else if (MODE == 1) {
          ((float*)C)[(size_t)row * N + col] = v;
        } else {
          int b = row >> 11, s = row & (SEQ - 1);
          int hk = col / HD, d = col % HD;
          ((_Float16*)C)[(((size_t)(b * HKV + hk)) * DPAD + d) * SEQ + s] =
              (_Float16)v;
        }
      }
    }
  }
}

// ---------------- Flash attention (causal), 1 wave per 16-query tile ----------
// Vt column d==56 is all-ones, so the PV accumulator o[3] carries the softmax
// denominator l per row (rescaled by alpha exactly like the output is).
__global__ __launch_bounds__(32) void attn_flash(
    const _Float16* __restrict__ Qh,  // [ROWS][EMBED]
    const _Float16* __restrict__ Kh,  // [ROWS][KV_EMBED]
    const _Float16* __restrict__ Vt,  // [B][HKV][DPAD][SEQ] (d-padded)
    float* __restrict__ Oattn,        // [ROWS][EMBED]
    float* __restrict__ Mst,          // [B][HQ][SEQ]
    float* __restrict__ Lst) {
  __shared__ _Float16 Plds[16 * 32];
  const int lane = threadIdx.x, lm = lane & 15, lh = lane >> 4;
  const int q0 = blockIdx.x * 16;
  const int h = blockIdx.y;
  const int b = blockIdx.z;
  const int hk = h >> 2;

  const _Float16* qrow = Qh + (size_t)(b * SEQ + q0 + lm) * EMBED + h * HD;
  v16h aq0 = afrag_f16b(qrow, lh, 0, HD);
  v16h aq1 = afrag_f16b(qrow, lh, 32, HD);

  float m_run[8];
  v8f zero = {};
  v8f o[4];
#pragma unroll
  for (int t = 0; t < 4; ++t) o[t] = zero;
#pragma unroll
  for (int r = 0; r < 8; ++r) m_run[r] = -3.0e38f;

  auto kv_step = [&](int s0, bool masked) {
    // ---- scores: 2 column tiles of 16 kv, contraction over d (64, padded)
    const _Float16* kr0 = Kh + (size_t)(b * SEQ + s0 + lm) * KV_EMBED + hk * HD;
    const _Float16* kr1 = kr0 + (size_t)16 * KV_EMBED;
    v8f s0c = zero, s1c = zero;
    {
      v16h bk = bfrag_f16b(kr0, lh, 0, HD);
      s0c = wmma_f16(aq0, bk, s0c);
      bk = bfrag_f16b(kr0, lh, 32, HD);
      s0c = wmma_f16(aq1, bk, s0c);
      bk = bfrag_f16b(kr1, lh, 0, HD);
      s1c = wmma_f16(aq0, bk, s1c);
      bk = bfrag_f16b(kr1, lh, 32, HD);
      s1c = wmma_f16(aq1, bk, s1c);
    }
    float x0[8], x1[8];
#pragma unroll
    for (int r = 0; r < 8; ++r) {
      x0[r] = s0c[r] * QK_SCALE;
      x1[r] = s1c[r] * QK_SCALE;
    }
    if (masked) {  // only the diagonal-straddling block pays for the mask
#pragma unroll
      for (int r = 0; r < 8; ++r) {
        int qi = q0 + r + 8 * lh;
        if (s0 + lm > qi) x0[r] = -3.0e38f;
        if (s0 + 16 + lm > qi) x1[r] = -3.0e38f;
      }
    }
    // ---- online softmax max + rescale (sum is fused into PV via ones-column)
    float p0[8], p1[8], alpha[8];
#pragma unroll
    for (int r = 0; r < 8; ++r) {
      float mx = fmaxf(x0[r], x1[r]);
#pragma unroll
      for (int off = 8; off >= 1; off >>= 1) mx = fmaxf(mx, __shfl_xor(mx, off));
      float nm = fmaxf(m_run[r], mx);
      alpha[r] = __expf(m_run[r] - nm);
      m_run[r] = nm;
      p0[r] = __expf(x0[r] - nm);
      p1[r] = __expf(x1[r] - nm);
    }
#pragma unroll
    for (int t = 0; t < 4; ++t)
#pragma unroll
      for (int r = 0; r < 8; ++r) o[t][r] *= alpha[r];

    // ---- transpose P (D layout -> A layout) through LDS
    __syncthreads();  // protect previous iteration's reads
#pragma unroll
    for (int r = 0; r < 8; ++r) {
      Plds[(r + 8 * lh) * 32 + lm] = (_Float16)p0[r];
      Plds[(r + 8 * lh) * 32 + 16 + lm] = (_Float16)p1[r];
    }
    __syncthreads();
    v16h pa;
#pragma unroll
    for (int j = 0; j < 8; ++j) {
      int k = ka_of(j, lh);
      pa[2 * j] = Plds[lm * 32 + k];
      pa[2 * j + 1] = Plds[lm * 32 + k + 1];
    }
    // ---- PV: contraction over 32 kv, 4 d-tiles of 16 (tile 3 holds l at d=56)
#pragma unroll
    for (int t = 0; t < 4; ++t) {
      const _Float16* vrow =
          Vt + ((size_t)(b * HKV + hk) * DPAD + t * 16 + lm) * SEQ + s0;
      v16h bv;
#pragma unroll
      for (int j = 0; j < 8; ++j) {
        int k = kb_of(j, lh);
        h16x2 d = *reinterpret_cast<const h16x2*>(vrow + k);
        bv[2 * j] = d.x;
        bv[2 * j + 1] = d.y;
      }
      o[t] = wmma_f16(pa, bv, o[t]);
    }
  };

  int s0 = 0;
  for (; s0 + 31 <= q0; s0 += 32) kv_step(s0, false);  // fully-unmasked blocks
  for (; s0 < q0 + 16; s0 += 32) kv_step(s0, true);    // diagonal block(s)

  // ---- epilogue: l lives in o[3] at column d==56 (lane lm==8 of each half)
  float l_run[8], inv_l[8];
#pragma unroll
  for (int r = 0; r < 8; ++r) {
    l_run[r] = __shfl(o[3][r], 8 + 16 * lh);  // broadcast within lane half
    inv_l[r] = 1.0f / l_run[r];
  }
#pragma unroll
  for (int t = 0; t < 4; ++t) {
#pragma unroll
    for (int r = 0; r < 8; ++r) {
      int d = t * 16 + lm;
      if (d < HD)
        Oattn[(size_t)(b * SEQ + q0 + r + 8 * lh) * EMBED + h * HD + d] =
            o[t][r] * inv_l[r];
    }
  }
#pragma unroll
  for (int r = 0; r < 8; ++r) {
    if (lm == r) {
      size_t idx = ((size_t)(b * HQ + h)) * SEQ + q0 + r + 8 * lh;
      Mst[idx] = m_run[r];
      Lst[idx] = l_run[r];
    }
  }
}

// ---------------- averaged attention weights: aw[b][s][h] = mean_n attn[n][s]
__global__ __launch_bounds__(32) void attn_weights(
    const _Float16* __restrict__ Qh, const _Float16* __restrict__ Kh,
    const float* __restrict__ Mst, const float* __restrict__ Lst,
    float* __restrict__ aw) {
  const int lane = threadIdx.x, lm = lane & 15, lh = lane >> 4;
  const int s0 = blockIdx.x * 16;
  const int h = blockIdx.y;
  const int b = blockIdx.z;
  const int hk = h >> 2;

  const _Float16* kr = Kh + (size_t)(b * SEQ + s0 + lm) * KV_EMBED + hk * HD;
  v16h bk0 = bfrag_f16b(kr, lh, 0, HD);
  v16h bk1 = bfrag_f16b(kr, lh, 32, HD);
  const float* Mrow = Mst + ((size_t)(b * HQ + h)) * SEQ;
  const float* Lrow = Lst + ((size_t)(b * HQ + h)) * SEQ;
  v8f zero = {};
  float acc = 0.f;

  auto score_tile = [&](int n0) -> v8f {
    const _Float16* qrow = Qh + (size_t)(b * SEQ + n0 + lm) * EMBED + h * HD;
    v16h a0 = afrag_f16b(qrow, lh, 0, HD);
    v16h a1 = afrag_f16b(qrow, lh, 32, HD);
    v8f c = zero;
    c = wmma_f16(a0, bk0, c);
    c = wmma_f16(a1, bk1, c);
    return c;
  };

  {  // diagonal block n0 == s0: needs the causal guard
    v8f c = score_tile(s0);
#pragma unroll
    for (int r = 0; r < 8; ++r) {
      int n = s0 + r + 8 * lh;
      if (s0 + lm <= n)
        acc += __expf(c[r] * QK_SCALE - Mrow[n]) * (1.0f / Lrow[n]);
    }
  }
  for (int n0 = s0 + 16; n0 < SEQ; n0 += 16) {  // n >= s guaranteed: no guard
    v8f c = score_tile(n0);
#pragma unroll
    for (int r = 0; r < 8; ++r) {
      int n = n0 + r + 8 * lh;
      acc += __expf(c[r] * QK_SCALE - Mrow[n]) * (1.0f / Lrow[n]);
    }
  }
  acc += __shfl_xor(acc, 16);  // combine lane halves (rows r and r+8)
  if (lh == 0)
    aw[((size_t)(b * SEQ) + s0 + lm) * HQ + h] = acc * (1.0f / (float)SEQ);
}

// ---------------- LayerNorm over last dim (896) ----------------
__global__ __launch_bounds__(256) void layernorm_k(const float* __restrict__ x,
                                                   const float* __restrict__ g,
                                                   const float* __restrict__ be,
                                                   float* __restrict__ y) {
  const int row = blockIdx.x;
  const int tid = threadIdx.x;
  const float* xr = x + (size_t)row * EMBED;
  float s = 0.f, ss = 0.f;
  for (int i = tid; i < EMBED; i += 256) {
    float v = xr[i];
    s += v;
    ss += v * v;
  }
  __shared__ float sh[512];
  sh[tid] = s;
  sh[256 + tid] = ss;
  __syncthreads();
  for (int st = 128; st > 0; st >>= 1) {
    if (tid < st) {
      sh[tid] += sh[tid + st];
      sh[256 + tid] += sh[256 + tid + st];
    }
    __syncthreads();
  }
  float mu = sh[0] * (1.0f / EMBED);
  float var = sh[256] * (1.0f / EMBED) - mu * mu;
  float rstd = rsqrtf(var + LN_EPS);
  for (int i = tid; i < EMBED; i += 256) {
    y[(size_t)row * EMBED + i] = (xr[i] - mu) * rstd * g[i] + be[i];
  }
}

// Vt pad-fill: zeros everywhere except d==56 which is 1.0 (ones-column that
// makes the PV WMMA accumulate the softmax denominator).
__global__ void fill_vt_pad(unsigned int* p, int n) {
  int i = blockIdx.x * 256 + threadIdx.x;
  if (i < n) {
    int d = (i >> 10) & 63;  // 1024 u32 (2048 f16) per d-row
    p[i] = (d == 56) ? 0x3C003C00u : 0u;
  }
}

extern "C" void kernel_launch(void* const* d_in, const int* in_sizes, int n_in,
                              void* d_out, int out_size, void* d_ws,
                              size_t ws_size, hipStream_t stream) {
  const float* q = (const float*)d_in[0];
  const float* k = (const float*)d_in[1];
  const float* v = (const float*)d_in[2];
  const float* Wq = (const float*)d_in[3];
  const float* Wk = (const float*)d_in[4];
  const float* Wv = (const float*)d_in[5];
  const float* Wo = (const float*)d_in[6];
  const float* gam = (const float*)d_in[7];
  const float* bet = (const float*)d_in[8];

  // workspace layout (bytes), ~41.2 MB total — fits comfortably in L2 (192 MB)
  char* w = (char*)d_ws;
  _Float16* Qh = (_Float16*)(w);                // 4096*896*2  = 7,340,032
  _Float16* Kh = (_Float16*)(w + 7340032);      // 4096*224*2  = 1,835,008
  _Float16* Vt = (_Float16*)(w + 9175040);      // 2*4*64*2048*2 = 2,097,152
  float* Mst = (float*)(w + 11272192);          // 2*16*2048*4 = 262,144
  float* Lst = (float*)(w + 11534336);          // 262,144
  float* Oattn = (float*)(w + 11796480);        // 4096*896*4  = 14,680,064
  float* LNout = (float*)(w + 26476544);        // 14,680,064
  float* outp = (float*)d_out;
  float* awp = outp + (size_t)ROWS * EMBED;

  // pad-fill Vt (d 57..63 zero, d==56 ones; valid d<56 written by gemm)
  fill_vt_pad<<<(524288 + 255) / 256, 256, 0, stream>>>((unsigned int*)Vt,
                                                        524288);
  // projections (f32 in, f16 out)
  gemm16<0, 4><<<dim3(EMBED / 64, ROWS / 32), 32, 0, stream>>>(q, Wq, Qh, ROWS,
                                                               EMBED, EMBED);
  gemm16<0, 2><<<dim3(KV_EMBED / 32, ROWS / 32), 32, 0, stream>>>(
      k, Wk, Kh, ROWS, KV_EMBED, EMBED);
  gemm16<2, 2><<<dim3(KV_EMBED / 32, ROWS / 32), 32, 0, stream>>>(
      v, Wv, Vt, ROWS, KV_EMBED, EMBED);
  // flash attention + stats
  attn_flash<<<dim3(SEQ / 16, HQ, BATCH), 32, 0, stream>>>(Qh, Kh, Vt, Oattn,
                                                           Mst, Lst);
  // averaged attention weights (second tail of d_out)
  attn_weights<<<dim3(SEQ / 16, HQ, BATCH), 32, 0, stream>>>(Qh, Kh, Mst, Lst,
                                                             awp);
  // layernorm + output projection (f32 out straight into d_out)
  layernorm_k<<<ROWS, 256, 0, stream>>>(Oattn, gam, bet, LNout);
  gemm16<1, 4><<<dim3(EMBED / 64, ROWS / 32), 32, 0, stream>>>(
      LNout, Wo, outp, ROWS, EMBED, EMBED);
}
